// AttentionBlock_29257317220724
// MI455X (gfx1250) — matrix operand order repaired and checked by reading, not compiled, not run on hardware
//
#include <hip/hip_runtime.h>

typedef __attribute__((ext_vector_type(16))) _Float16 v16h;
typedef __attribute__((ext_vector_type(8)))  _Float16 v8h;
typedef __attribute__((ext_vector_type(8)))  float    v8f;

#define BDIM   2
#define CDIM   256
#define NTOK   4096
#define NGRP   8
#define NHEADS 4
#define HD     64

// ---------------------------------------------------------------------------
// Fragment helpers using a contiguous k order for A and B, not the CDNA5 table order (CDNA5 ISA)
//
// A-matrix 16x32 (MxK), row M = lane%16:
//   element i<8  : K = koff + i           (koff = (lane&16)?16:0)
//   element i>=8 : K = koff + i           (i.e. K in [koff+8, koff+16))
// -> two contiguous 8-element chunks per lane.
// B-matrix 32x16 (KxN): N = lane%16, element i -> K = ((lane&16)?16:0) + i
// -> one contiguous 16-element (32B) chunk per lane.
__device__ __forceinline__ v16h afrag_from_f16(const _Float16* p0, const _Float16* p1) {
    v8h a = *(const v8h*)p0;
    v8h b = *(const v8h*)p1;
    v16h f;
#pragma unroll
    for (int i = 0; i < 8; ++i) { f[i] = a[i]; f[i + 8] = b[i]; }
    return f;
}

// ---------------------------------------------------------------------------
// 0) One-shot fp32 -> f16 weight conversion (removes per-tile cvt chains).
__global__ void cvt_w_kernel(const float* __restrict__ src, _Float16* __restrict__ dst, int n) {
    const int i = blockIdx.x * 256 + threadIdx.x;
    if (i < n) dst[i] = (_Float16)src[i];
}

// ---------------------------------------------------------------------------
// 1) GroupNorm statistics: one block per (b, g), reduce 32*4096 elements.
__global__ void gn_stats_kernel(const float* __restrict__ x, float* __restrict__ stats) {
    const int bg = blockIdx.x;                               // b*NGRP + g
    const size_t base = (size_t)bg * (CDIM / NGRP) * NTOK;   // groups contiguous in C
    const int n = (CDIM / NGRP) * NTOK;                      // 131072
    float s = 0.f, s2 = 0.f;
    for (int i = threadIdx.x; i < n; i += 256) {
        float v = x[base + i];
        s += v; s2 += v * v;
    }
    __shared__ float red[256], red2[256];
    red[threadIdx.x] = s; red2[threadIdx.x] = s2;
    __syncthreads();
#pragma unroll
    for (int off = 128; off > 0; off >>= 1) {
        if (threadIdx.x < off) {
            red[threadIdx.x]  += red[threadIdx.x + off];
            red2[threadIdx.x] += red2[threadIdx.x + off];
        }
        __syncthreads();
    }
    if (threadIdx.x == 0) {
        float mean = red[0] / (float)n;
        float var  = red2[0] / (float)n - mean * mean;
        stats[2 * bg]     = mean;
        stats[2 * bg + 1] = rsqrtf(var + 1e-5f);
    }
}

// ---------------------------------------------------------------------------
// 2) Apply GroupNorm + affine, transpose to token-major f16: xn_t[b][n][c].
__global__ void gn_apply_kernel(const float* __restrict__ x,
                                const float* __restrict__ gw,
                                const float* __restrict__ gb,
                                const float* __restrict__ stats,
                                _Float16* __restrict__ xn_t) {
    const int b = blockIdx.x >> 4;
    const int n = ((blockIdx.x & 15) << 8) + threadIdx.x;    // coalesced in n
    const float* xb = x + (size_t)b * CDIM * NTOK + n;
    _Float16* dst = xn_t + ((size_t)b * NTOK + n) * CDIM;
#pragma unroll 1
    for (int g = 0; g < NGRP; ++g) {
        const float mean = stats[2 * (b * NGRP + g)];
        const float rstd = stats[2 * (b * NGRP + g) + 1];
#pragma unroll 4
        for (int ci = 0; ci < CDIM / NGRP; ++ci) {
            const int c = g * (CDIM / NGRP) + ci;
            const float v = (xb[(size_t)c * NTOK] - mean) * rstd * gw[c] + gb[c];
            dst[c] = (_Float16)v;
        }
    }
}

// ---------------------------------------------------------------------------
// 3) QKV GEMM: out[o,n] = qkv_w[o,:] . xn[:,n] + qkv_b[o].
//    One wave per 16x16 tile, K = 256 in 8 WMMA steps; f16 weights.
//    q scaled by hd^-0.5 = 0.125 at store; v stored transposed [b][h][d][n].
__global__ void __launch_bounds__(256) qkv_kernel(const _Float16* __restrict__ w16,
                                                  const float* __restrict__ bias,
                                                  const _Float16* __restrict__ xn_t,
                                                  _Float16* __restrict__ q,
                                                  _Float16* __restrict__ kbuf,
                                                  _Float16* __restrict__ vt) {
    const int lane = threadIdx.x & 31;
    const int w_id = (blockIdx.x << 3) + (threadIdx.x >> 5);
    const int b  = w_id / (48 * 256);
    const int rm = w_id % (48 * 256);
    const int ot = rm >> 8;                                  // 0..47 (rows of 3C)
    const int nt = rm & 255;                                 // 0..255 (token tiles)
    const int half8 = (lane & 16) ? 8 : 0;
    const int koff  = (lane & 16) ? 16 : 0;
    const _Float16* arow = w16 + (size_t)(ot * 16 + (lane & 15)) * CDIM;
    const _Float16* brow = xn_t + ((size_t)b * NTOK + nt * 16 + (lane & 15)) * CDIM;

    v8f acc = {};
#pragma unroll
    for (int k0 = 0; k0 < CDIM; k0 += 32) {
        v16h af = afrag_from_f16(arow + k0 + koff, arow + k0 + koff + 8);
        v16h bf = *(const v16h*)(brow + k0 + koff);
        acc = __builtin_amdgcn_wmma_f32_16x16x32_f16(false, af, false, bf,
                                                     (short)0, acc, false, false);
    }

    const int n = nt * 16 + (lane & 15);
#pragma unroll
    for (int r = 0; r < 8; ++r) {
        const int o   = ot * 16 + half8 + r;                 // C-tile row M
        const float v = acc[r] + bias[o];
        const int s = o >> 8, h = (o >> 6) & 3, d = o & 63;
        const size_t bh = (size_t)(b * NHEADS + h);
        if (s == 0)      q   [(bh * NTOK + n) * HD + d] = (_Float16)(v * 0.125f);
        else if (s == 1) kbuf[(bh * NTOK + n) * HD + d] = (_Float16)v;
        else             vt  [(bh * HD + d) * NTOK + n] = (_Float16)v;
    }
}

// ---------------------------------------------------------------------------
// 4) Flash attention. One block = 8 waves sharing one (b, h); each wave owns a
//    16-query tile. Per 32-key step, the K tile (32x64) and V^T tile (64x32)
//    are staged once per block into double-buffered LDS with CDNA5 async
//    global->LDS loads (ASYNCcnt), then all 8 waves build fragments from LDS.
__global__ void __launch_bounds__(256) attn_kernel(const _Float16* __restrict__ q,
                                                   const _Float16* __restrict__ kk,
                                                   const _Float16* __restrict__ vt,
                                                   _Float16* __restrict__ h_t) {
    __shared__ __align__(32) _Float16 kbuf[2][32 * 64];      // 2 x 4 KB
    __shared__ __align__(32) _Float16 vbuf[2][64 * 32];      // 2 x 4 KB
    __shared__ __align__(32) _Float16 plds[8][16 * 32];      // per-wave P tile, 8 KB

    const int tid  = threadIdx.x;
    const int lane = tid & 31;
    const int wv   = tid >> 5;
    const int b    = blockIdx.x >> 7;                        // 256 blocks total
    const int hh   = (blockIdx.x >> 5) & 3;
    const int qt   = ((blockIdx.x & 31) << 3) + wv;
    const int half8 = (lane & 16) ? 8 : 0;
    const int koff  = (lane & 16) ? 16 : 0;
    const size_t bh = (size_t)(b * NHEADS + hh);

    const _Float16* qrow  = q  + (bh * NTOK + qt * 16 + (lane & 15)) * HD;
    const _Float16* kbase = kk + bh * NTOK * HD;
    const _Float16* vbase = vt + bh * HD * NTOK;

    // Per-thread 16B chunk coordinates for the cooperative async tile loads.
    const int krow = tid >> 3, kcol = (tid & 7) * 8;         // K tile 32x64
    const int vrow = tid >> 2, vcol = (tid & 3) * 8;         // V^T tile 64x32

    auto issue_tile = [&](int key0, int bufi) {
        const _Float16* gk = kbase + (size_t)(key0 + krow) * HD + kcol;
        const _Float16* gv = vbase + (size_t)vrow * NTOK + key0 + vcol;
        // Low 32 bits of a generic pointer to __shared__ = LDS byte offset.
        unsigned lk = (unsigned)(uintptr_t)&kbuf[bufi][krow * 64 + kcol];
        unsigned lv = (unsigned)(uintptr_t)&vbuf[bufi][vrow * 32 + vcol];
        asm volatile("global_load_async_to_lds_b128 %0, %1, off\n\t"
                     "global_load_async_to_lds_b128 %2, %3, off"
                     :: "v"(lk), "v"(gk), "v"(lv), "v"(gv)
                     : "memory");
    };

    // Q fragments for d=[0,32) and d=[32,64)   (q already scaled by 0.125)
    const v16h qf0 = afrag_from_f16(qrow + koff,      qrow + koff + 8);
    const v16h qf1 = afrag_from_f16(qrow + 32 + koff, qrow + 32 + koff + 8);

    float mrow[8], lrow[8];
    v8f oacc[4];
#pragma unroll
    for (int r = 0; r < 8; ++r) { mrow[r] = -3.0e38f; lrow[r] = 0.f; }
#pragma unroll
    for (int d = 0; d < 4; ++d) { v8f z = {}; oacc[d] = z; }

    _Float16* pw = plds[wv];
    const _Float16* prd = pw + (lane & 15) * 32 + koff;      // A-frag read base

    issue_tile(0, 0);

    for (int kt = 0; kt < NTOK / 32; ++kt) {
        const int bufi = kt & 1;
        const int key0 = kt * 32;
        asm volatile("s_wait_asynccnt 0x0" ::: "memory");    // my async loads done
        __syncthreads();                                     // everyone's done; prev buf free
        if (kt + 1 < NTOK / 32) issue_tile(key0 + 32, bufi ^ 1);

        const _Float16* kr0 = kbuf[bufi] + (lane & 15) * 64;
        const _Float16* kr1 = kbuf[bufi] + (16 + (lane & 15)) * 64;
        const v16h kf00 = *(const v16h*)(kr0 + koff);
        const v16h kf01 = *(const v16h*)(kr0 + 32 + koff);
        const v16h kf10 = *(const v16h*)(kr1 + koff);
        const v16h kf11 = *(const v16h*)(kr1 + 32 + koff);

        v8f s0 = {}, s1 = {};
        s0 = __builtin_amdgcn_wmma_f32_16x16x32_f16(false, qf0, false, kf00, (short)0, s0, false, false);
        s0 = __builtin_amdgcn_wmma_f32_16x16x32_f16(false, qf1, false, kf01, (short)0, s0, false, false);
        s1 = __builtin_amdgcn_wmma_f32_16x16x32_f16(false, qf0, false, kf10, (short)0, s1, false, false);
        s1 = __builtin_amdgcn_wmma_f32_16x16x32_f16(false, qf1, false, kf11, (short)0, s1, false, false);

        // Online softmax: each lane-half owns rows M = half8 + r of the C tile.
#pragma unroll
        for (int r = 0; r < 8; ++r) {
            float t = fmaxf(s0[r], s1[r]);
#pragma unroll
            for (int off = 1; off < 16; off <<= 1) t = fmaxf(t, __shfl_xor(t, off));
            const float mnew = fmaxf(mrow[r], t);
            const float corr = __expf(mrow[r] - mnew);
            mrow[r] = mnew;
            const float p0 = __expf(s0[r] - mnew);
            const float p1 = __expf(s1[r] - mnew);
            float rs = p0 + p1;
#pragma unroll
            for (int off = 1; off < 16; off <<= 1) rs += __shfl_xor(rs, off);
            lrow[r] = lrow[r] * corr + rs;
#pragma unroll
            for (int d = 0; d < 4; ++d) oacc[d][r] *= corr;
            const int m = half8 + r;
            pw[m * 32 + (lane & 15)]      = (_Float16)p0;    // cols  0..15
            pw[m * 32 + 16 + (lane & 15)] = (_Float16)p1;    // cols 16..31
        }
        // LDS ops are issued in order within a wave; fence compiler reordering.
        asm volatile("" ::: "memory");
        const v16h pf = afrag_from_f16(prd, prd + 8);        // P as 16x32 A-frag
        asm volatile("" ::: "memory");

#pragma unroll
        for (int d = 0; d < 4; ++d) {
            const _Float16* vr = vbuf[bufi] + (d * 16 + (lane & 15)) * 32 + koff;
            const v16h vf = *(const v16h*)vr;
            oacc[d] = __builtin_amdgcn_wmma_f32_16x16x32_f16(false, pf, false, vf,
                                                             (short)0, oacc[d], false, false);
        }
    }

    // Normalize and store h token-major: h_t[b][n][c], c = hh*64 + d.
#pragma unroll
    for (int r = 0; r < 8; ++r) {
        const float inv = 1.f / lrow[r];
        const int nq = qt * 16 + half8 + r;
        _Float16* dst = h_t + ((size_t)b * NTOK + nq) * CDIM + hh * HD;
#pragma unroll
        for (int d = 0; d < 4; ++d)
            dst[d * 16 + (lane & 15)] = (_Float16)(oacc[d][r] * inv);
    }
}

// ---------------------------------------------------------------------------
// 5) Output projection + bias + residual (fp32 out); f16 weights.
__global__ void __launch_bounds__(256) proj_kernel(const _Float16* __restrict__ w16,
                                                   const float* __restrict__ bias,
                                                   const _Float16* __restrict__ h_t,
                                                   const float* __restrict__ x,
                                                   float* __restrict__ out) {
    const int lane = threadIdx.x & 31;
    const int w_id = (blockIdx.x << 3) + (threadIdx.x >> 5);
    const int b  = w_id >> 12;
    const int ot = (w_id >> 8) & 15;
    const int nt = w_id & 255;
    const int half8 = (lane & 16) ? 8 : 0;
    const int koff  = (lane & 16) ? 16 : 0;
    const _Float16* arow = w16 + (size_t)(ot * 16 + (lane & 15)) * CDIM;
    const _Float16* brow = h_t + ((size_t)b * NTOK + nt * 16 + (lane & 15)) * CDIM;

    v8f acc = {};
#pragma unroll
    for (int k0 = 0; k0 < CDIM; k0 += 32) {
        v16h af = afrag_from_f16(arow + k0 + koff, arow + k0 + koff + 8);
        v16h bf = *(const v16h*)(brow + k0 + koff);
        acc = __builtin_amdgcn_wmma_f32_16x16x32_f16(false, af, false, bf,
                                                     (short)0, acc, false, false);
    }

    const int n = nt * 16 + (lane & 15);
#pragma unroll
    for (int r = 0; r < 8; ++r) {
        const int o = ot * 16 + half8 + r;
        const size_t idx = ((size_t)b * CDIM + o) * NTOK + n;
        out[idx] = acc[r] + bias[o] + x[idx];
    }
}

// ---------------------------------------------------------------------------
extern "C" void kernel_launch(void* const* d_in, const int* in_sizes, int n_in,
                              void* d_out, int out_size, void* d_ws, size_t ws_size,
                              hipStream_t stream) {
    const float* x      = (const float*)d_in[0];
    const float* norm_w = (const float*)d_in[1];
    const float* norm_b = (const float*)d_in[2];
    const float* qkv_w  = (const float*)d_in[3];
    const float* qkv_b  = (const float*)d_in[4];
    const float* proj_w = (const float*)d_in[5];
    const float* proj_b = (const float*)d_in[6];
    float* out = (float*)d_out;

    // Workspace carve-up (256B aligned; ~20.5 MiB total).
    char* ws = (char*)d_ws;
    float* stats = (float*)ws;
    size_t off = 256;
    const size_t sz_tok = (size_t)BDIM * NTOK * CDIM * sizeof(_Float16);      // 4 MiB
    _Float16* xn_t = (_Float16*)(ws + off); off += sz_tok;
    _Float16* qb   = (_Float16*)(ws + off); off += sz_tok;
    _Float16* kb   = (_Float16*)(ws + off); off += sz_tok;
    _Float16* vtb  = (_Float16*)(ws + off); off += sz_tok;
    _Float16* h_t  = (_Float16*)(ws + off); off += sz_tok;
    _Float16* wq16 = (_Float16*)(ws + off); off += (size_t)3 * CDIM * CDIM * 2; // 384 KiB
    _Float16* wp16 = (_Float16*)(ws + off); off += (size_t)CDIM * CDIM * 2;     // 128 KiB

    cvt_w_kernel<<<(3 * CDIM * CDIM) / 256, 256, 0, stream>>>(qkv_w, wq16, 3 * CDIM * CDIM);
    cvt_w_kernel<<<(CDIM * CDIM) / 256, 256, 0, stream>>>(proj_w, wp16, CDIM * CDIM);
    gn_stats_kernel<<<BDIM * NGRP, 256, 0, stream>>>(x, stats);
    gn_apply_kernel<<<BDIM * 16, 256, 0, stream>>>(x, norm_w, norm_b, stats, xn_t);
    qkv_kernel<<<(BDIM * 48 * 256) / 8, 256, 0, stream>>>(wq16, qkv_b, xn_t, qb, kb, vtb);
    attn_kernel<<<BDIM * NHEADS * 32, 256, 0, stream>>>(qb, kb, vtb, h_t);
    proj_kernel<<<(BDIM * 16 * 256) / 8, 256, 0, stream>>>(wp16, proj_b, h_t, x, out);
}
